// LinearSigmoid_18_18442589569806
// MI455X (gfx1250) — matrix operand order repaired and checked by reading, not compiled, not run on hardware
//
#include <hip/hip_runtime.h>
#include <hip/hip_bf16.h>

// ---------------------------------------------------------------------------
// Fused MLP for MI455X (gfx1250, wave32):
//   h1 = sigmoid(x @ W0^T)        : 32768x784 @ 784x39  -> bf16 WMMA (bulk)
//   h2 = sigmoid(h1 @ W2^T + b2)  : tiny -> per-row VALU from LDS
//   h3 = sigmoid(h2 @ W4^T)
//   h4 = sigmoid(h3 @ W6^T + b6)
//   out = log_softmax(h4 @ W8^T)
// ---------------------------------------------------------------------------

typedef __attribute__((ext_vector_type(16))) __bf16 v16bf;
typedef __attribute__((ext_vector_type(8)))  float  v8f;
typedef __attribute__((ext_vector_type(8)))  __bf16 v8bf;
typedef __attribute__((ext_vector_type(4)))  float  f32x4;

static_assert(sizeof(v16bf) == 32, "v16bf must be 32 bytes");

#define MLP_B     32768
#define MLP_K0    784
#define MLP_K0P   800   // 25 * 32 (WMMA K steps)
#define MLP_N0    39
#define MLP_N0P   48    // 3 * 16 (WMMA N tiles)
#define ROWS_PER_BLOCK 128   // 8 waves x 16 rows

// --- pre-kernel: pack W0 [39][784] f32 -> zero-padded bf16 [48][800] --------
__global__ void w0_pack_kernel(const float* __restrict__ W0,
                               __bf16* __restrict__ w0b) {
    int idx = blockIdx.x * 256 + threadIdx.x;
    if (idx >= MLP_N0P * MLP_K0P) return;
    int n = idx / MLP_K0P;
    int k = idx % MLP_K0P;
    float v = (n < MLP_N0 && k < MLP_K0) ? W0[n * MLP_K0 + k] : 0.0f;
    w0b[idx] = (__bf16)v;
}

__device__ __forceinline__ float sigmoidf_(float v) {
    return 1.0f / (1.0f + __expf(-v));
}

// --- main fused kernel ------------------------------------------------------
__global__ void __launch_bounds__(256, 1)
mlp_fused_kernel(const float*  __restrict__ x,
                 const __bf16* __restrict__ w0b,
                 const float*  __restrict__ W2, const float* __restrict__ b2,
                 const float*  __restrict__ W4, const float* __restrict__ W6,
                 const float*  __restrict__ b6, const float* __restrict__ W8,
                 float* __restrict__ out) {
    __shared__ float lds_h1[ROWS_PER_BLOCK][MLP_N0P];   // 24 KB

    const int tid  = threadIdx.x;
    const int wave = tid >> 5;
    const int lane = tid & 31;
    const int hl   = lane >> 4;   // half-wave id (0/1)
    const int m16  = lane & 15;

    const int rowBase = blockIdx.x * ROWS_PER_BLOCK + wave * 16;
    const float* xr = x + (size_t)(rowBase + m16) * MLP_K0;

    v8f acc0 = {0.f,0.f,0.f,0.f,0.f,0.f,0.f,0.f};
    v8f acc1 = {0.f,0.f,0.f,0.f,0.f,0.f,0.f,0.f};
    v8f acc2 = {0.f,0.f,0.f,0.f,0.f,0.f,0.f,0.f};

    // ---- layer 0: 25 K-steps of v_wmma_f32_16x16x32_bf16, N tiles 0..2 ----
    for (int ks = 0; ks < 25; ++ks) {
        const int k0 = ks * 32;
        const int ka = k0 + hl * 8;    // first 8-wide K chunk for this lane
        const int kb = ka + 16;        // second 8-wide K chunk

        // A fragment (16-bit A 16x32 layout): lane<16 -> row M=lane, K {0..7,16..23};
        // lane>=16 -> row M=lane-16, K {8..15,24..31}. Two contiguous 32B loads.
        f32x4 f0 = *(const f32x4*)(xr + ka);
        f32x4 f1 = *(const f32x4*)(xr + ka + 4);
        f32x4 f2 = {0.f,0.f,0.f,0.f};
        f32x4 f3 = {0.f,0.f,0.f,0.f};
        if (kb + 7 < MLP_K0) {         // uniform across the wave for every ks
            f2 = *(const f32x4*)(xr + kb);
            f3 = *(const f32x4*)(xr + kb + 4);
        }
        v16bf a;
        a[0]=(__bf16)f0.x;  a[1]=(__bf16)f0.y;  a[2]=(__bf16)f0.z;  a[3]=(__bf16)f0.w;
        a[4]=(__bf16)f1.x;  a[5]=(__bf16)f1.y;  a[6]=(__bf16)f1.z;  a[7]=(__bf16)f1.w;
        a[8]=(__bf16)f2.x;  a[9]=(__bf16)f2.y;  a[10]=(__bf16)f2.z; a[11]=(__bf16)f2.w;
        a[12]=(__bf16)f3.x; a[13]=(__bf16)f3.y; a[14]=(__bf16)f3.z; a[15]=(__bf16)f3.w;

        // B fragments (32x16 bf16): lanes 0-15 hold K {0..7,16..23} of column N=lane,
        // lanes 16-31 hold K {8..15,24..31} of column N=lane-16. Two 16B loads each.
        const __bf16* bbase = w0b + (size_t)m16 * MLP_K0P + k0 + hl * 8;
        auto ldb = [](const __bf16* p) {
            v8bf lo = *(const v8bf*)(p);
            v8bf hi = *(const v8bf*)(p + 16);
            return __builtin_shufflevector(lo, hi, 0,1,2,3,4,5,6,7,8,9,10,11,12,13,14,15);
        };
        v16bf bv0 = ldb(bbase);
        v16bf bv1 = ldb(bbase + 16 * MLP_K0P);
        v16bf bv2 = ldb(bbase + 32 * MLP_K0P);

        acc0 = __builtin_amdgcn_wmma_f32_16x16x32_bf16(false, a, false, bv0,
                                                       (short)0, acc0, false, false);
        acc1 = __builtin_amdgcn_wmma_f32_16x16x32_bf16(false, a, false, bv1,
                                                       (short)0, acc1, false, false);
        acc2 = __builtin_amdgcn_wmma_f32_16x16x32_bf16(false, a, false, bv2,
                                                       (short)0, acc2, false, false);
    }

    // ---- sigmoid in-register, stage h1 tile to LDS ----
    // C layout: VGPR r, lanes 0-15: (M=r, N=lane); lanes 16-31: (M=r+8, N=lane-16)
#pragma unroll
    for (int r = 0; r < 8; ++r) {
        const int m = wave * 16 + r + 8 * hl;
        lds_h1[m][m16]      = sigmoidf_(acc0[r]);
        lds_h1[m][16 + m16] = sigmoidf_(acc1[r]);
        lds_h1[m][32 + m16] = sigmoidf_(acc2[r]);
    }
    __syncthreads();

    // ---- layers 2..5 + log_softmax: one thread per row ----
    if (tid < ROWS_PER_BLOCK) {
        float h1[MLP_N0];
#pragma unroll
        for (int k = 0; k < MLP_N0; ++k) h1[k] = lds_h1[tid][k];

        float h2[26];
#pragma unroll
        for (int j = 0; j < 26; ++j) {
            float s = b2[j];
#pragma unroll
            for (int k = 0; k < 39; ++k) s += W2[j * 39 + k] * h1[k];
            h2[j] = sigmoidf_(s);
        }
        float h3[10];
#pragma unroll
        for (int j = 0; j < 10; ++j) {
            float s = 0.0f;
#pragma unroll
            for (int k = 0; k < 26; ++k) s += W4[j * 26 + k] * h2[k];
            h3[j] = sigmoidf_(s);
        }
        float h4[10];
#pragma unroll
        for (int j = 0; j < 10; ++j) {
            float s = b6[j];
#pragma unroll
            for (int k = 0; k < 10; ++k) s += W6[j * 10 + k] * h3[k];
            h4[j] = sigmoidf_(s);
        }
        float lg[10];
#pragma unroll
        for (int j = 0; j < 10; ++j) {
            float s = 0.0f;
#pragma unroll
            for (int k = 0; k < 10; ++k) s += W8[j * 10 + k] * h4[k];
            lg[j] = s;
        }
        float mx = lg[0];
#pragma unroll
        for (int j = 1; j < 10; ++j) mx = fmaxf(mx, lg[j]);
        float se = 0.0f;
#pragma unroll
        for (int j = 0; j < 10; ++j) se += __expf(lg[j] - mx);
        const float lse = __logf(se);

        const int row = blockIdx.x * ROWS_PER_BLOCK + tid;
        float* op = out + (size_t)row * 10;
#pragma unroll
        for (int j = 0; j < 10; ++j) op[j] = lg[j] - mx - lse;
    }
}

// ---------------------------------------------------------------------------
extern "C" void kernel_launch(void* const* d_in, const int* in_sizes, int n_in,
                              void* d_out, int out_size, void* d_ws, size_t ws_size,
                              hipStream_t stream) {
    (void)in_sizes; (void)n_in; (void)out_size; (void)ws_size;

    const float* x  = (const float*)d_in[0];
    const float* W0 = (const float*)d_in[1];
    const float* W2 = (const float*)d_in[2];
    const float* b2 = (const float*)d_in[3];
    const float* W4 = (const float*)d_in[4];
    const float* W6 = (const float*)d_in[5];
    const float* b6 = (const float*)d_in[6];
    const float* W8 = (const float*)d_in[7];
    float* out = (float*)d_out;

    __bf16* w0b = (__bf16*)d_ws;   // 48*800*2 = 76,800 bytes of scratch

    const int packElems  = MLP_N0P * MLP_K0P;
    const int packBlocks = (packElems + 255) / 256;
    w0_pack_kernel<<<packBlocks, 256, 0, stream>>>(W0, w0b);

    const int blocks = MLP_B / ROWS_PER_BLOCK;   // 256
    mlp_fused_kernel<<<blocks, 256, 0, stream>>>(x, w0b, W2, b2, W4, W6, b6, W8, out);
}
